// MoE_25065429139580
// MI455X (gfx1250) — compile-verified, matrix operand order checked by reading, not run on hardware
//
#include <hip/hip_runtime.h>
#include <hip/hip_bf16.h>

#define NUM_EXPERTS 8
#define TOPK 2
#define TOKENS 4096
#define HS 1024
#define FFN 4096
#define CAP 512
#define NBINS (NUM_EXPERTS * TOPK) // 16
#define LDA 40                     // 32 K-elems + 8 pad (80B rows, 16B-aligned chunks)

typedef __attribute__((ext_vector_type(16))) __bf16 v16bf;
typedef __attribute__((ext_vector_type(8)))  __bf16 bf16x8;
typedef __attribute__((ext_vector_type(8)))  float  v8f;
typedef __attribute__((ext_vector_type(4)))  int    v4i;

#if defined(__has_builtin)
#if __has_builtin(__builtin_amdgcn_global_load_async_to_lds_b128)
#define HAVE_ASYNC_LDS 1
#endif
#if __has_builtin(__builtin_amdgcn_s_wait_asynccnt)
#define HAVE_WAIT_ASYNC 1
#endif
#endif

__device__ __forceinline__ unsigned short f2bfu(float f) {
    unsigned u = __builtin_bit_cast(unsigned, f);
    unsigned r = (u + 0x7FFFu + ((u >> 16) & 1u)) >> 16; // round-to-nearest-even
    return (unsigned short)r;
}

__device__ __forceinline__ float gelu_tanh(float x) {
    float x3 = x * x * x;
    return 0.5f * x * (1.0f + tanhf(0.7978845608028654f * (x + 0.044715f * x3)));
}

// 16-byte global->LDS copy; async (memory->LDS DMA, ASYNCcnt) when available.
__device__ __forceinline__ void cp16_g2l(unsigned short* lds, const unsigned short* g) {
#if HAVE_ASYNC_LDS
    __builtin_amdgcn_global_load_async_to_lds_b128(
        (__attribute__((address_space(1))) v4i*)g,
        (__attribute__((address_space(3))) v4i*)lds, 0, 0);
#else
    *(bf16x8*)lds = *(const bf16x8*)g;
#endif
}

template <int N> __device__ __forceinline__ void wait_async() {
#if HAVE_WAIT_ASYNC
    __builtin_amdgcn_s_wait_asynccnt(N);
#else
    asm volatile("s_wait_asynccnt %0" ::"i"(N) : "memory");
#endif
}

__device__ __forceinline__ v8f wmma_bf16(v16bf a, v16bf b, v8f c) {
    return __builtin_amdgcn_wmma_f32_16x16x32_bf16(false, a, false, b, (short)0, c,
                                                   false, false);
}

// A/B fragment: lanes 0-15 -> row, K 0-7 & 16-23; lanes 16-31 -> K 8-15 & 24-31.
__device__ __forceinline__ v16bf make_frag(const unsigned short* lds, int row, int k0, int k1) {
    bf16x8 lo = *(const bf16x8*)(lds + row * LDA + k0);
    bf16x8 hi = *(const bf16x8*)(lds + row * LDA + k1);
    v16bf f;
#pragma unroll
    for (int i = 0; i < 8; ++i) { f[i] = lo[i]; f[8 + i] = hi[i]; }
    return f;
}

// ---------------- 0) Weight transpose+convert: [K][N] f32 -> [N][K] bf16 ----
__global__ __launch_bounds__(256)
void transpose_cvt(const float* __restrict__ src, unsigned short* __restrict__ dst,
                   int K, int N) {
    __shared__ float t[64][65];
    int e = blockIdx.z;
    src += (size_t)e * K * N;
    dst += (size_t)e * K * N;
    int k0 = blockIdx.y * 64, n0 = blockIdx.x * 64;
    int c = threadIdx.x & 63, r4 = threadIdx.x >> 6;
    for (int r = r4; r < 64; r += 4) t[r][c] = src[(size_t)(k0 + r) * N + n0 + c];
    __syncthreads();
    for (int r = r4; r < 64; r += 4) dst[(size_t)(n0 + r) * K + k0 + c] = f2bfu(t[c][r]);
}

// ---------------- 1) Router: softmax(x @ router_w) -> top-2 ----------------
__global__ __launch_bounds__(256)
void moe_router(const float* __restrict__ x, const float* __restrict__ rw,
                float* __restrict__ topw, int* __restrict__ tope) {
    int tok  = (blockIdx.x * blockDim.x + threadIdx.x) >> 5;
    int lane = threadIdx.x & 31;
    if (tok >= TOKENS) return;
    const float* xr = x + (size_t)tok * HS;
    float acc[NUM_EXPERTS];
#pragma unroll
    for (int e = 0; e < NUM_EXPERTS; ++e) acc[e] = 0.f;
    for (int h = lane; h < HS; h += 32) {
        float xv = xr[h];
#pragma unroll
        for (int e = 0; e < NUM_EXPERTS; ++e) acc[e] += xv * rw[h * NUM_EXPERTS + e];
    }
#pragma unroll
    for (int e = 0; e < NUM_EXPERTS; ++e)
#pragma unroll
        for (int off = 16; off > 0; off >>= 1)
            acc[e] += __shfl_xor(acc[e], off, 32);
    if (lane == 0) {
        float m = acc[0];
#pragma unroll
        for (int e = 1; e < NUM_EXPERTS; ++e) m = fmaxf(m, acc[e]);
        float p[NUM_EXPERTS], s = 0.f;
#pragma unroll
        for (int e = 0; e < NUM_EXPERTS; ++e) { p[e] = expf(acc[e] - m); s += p[e]; }
        float inv = 1.f / s;
#pragma unroll
        for (int e = 0; e < NUM_EXPERTS; ++e) p[e] *= inv;
        int e1 = 0; float b1 = p[0];
#pragma unroll
        for (int e = 1; e < NUM_EXPERTS; ++e) if (p[e] > b1) { b1 = p[e]; e1 = e; }
        int e2 = (e1 == 0) ? 1 : 0; float b2 = p[e2];
#pragma unroll
        for (int e = 0; e < NUM_EXPERTS; ++e)
            if (e != e1 && p[e] > b2) { b2 = p[e]; e2 = e; }
        topw[tok * TOPK + 0] = b1;  tope[tok * TOPK + 0] = e1;
        topw[tok * TOPK + 1] = b2;  tope[tok * TOPK + 1] = e2;
    }
}

// ---------------- 2) Stable counting sort into (k,expert) bins --------------
__global__ void moe_bin(const int* __restrict__ tope,
                        int* __restrict__ bins, int* __restrict__ counts) {
    int bin = blockIdx.x; // k*8 + e
    int k = bin >> 3, e = bin & 7;
    int lane = threadIdx.x; // blockDim.x == 32
    int base = 0;
    for (int t0 = 0; t0 < TOKENS; t0 += 32) {
        int t = t0 + lane;
        bool match = (tope[t * TOPK + k] == e);
        unsigned mm = (unsigned)__ballot(match);
        int slot = base + __popc(mm & ((1u << lane) - 1u));
        if (match && slot < CAP) bins[bin * CAP + slot] = t;
        base += __popc(mm);
    }
    if (lane == 0) counts[bin] = base < CAP ? base : CAP;
}

// ---------------- 3) Gather tokens into bf16 bins (zeros if invalid) --------
__global__ __launch_bounds__(256)
void moe_gather(const float* __restrict__ x, const int* __restrict__ bins,
                const int* __restrict__ counts, unsigned short* __restrict__ xe) {
    int bin  = blockIdx.x >> 9;
    int slot = blockIdx.x & (CAP - 1);
    size_t dst = ((size_t)bin * CAP + slot) * HS;
    if (slot < counts[bin]) {
        const float* src = x + (size_t)bins[bin * CAP + slot] * HS;
        for (int c = threadIdx.x; c < HS; c += 256) xe[dst + c] = f2bfu(src[c]);
    } else {
        for (int c = threadIdx.x; c < HS; c += 256) xe[dst + c] = 0;
    }
}

// ---------------- 4) out = bias broadcast -----------------------------------
__global__ __launch_bounds__(256)
void moe_bias_init(const float* __restrict__ bias, float* __restrict__ out) {
    int idx = blockIdx.x * 256 + threadIdx.x;
    if (idx < TOKENS * HS) out[idx] = bias[idx & (HS - 1)];
}

// ---------------- 5) h = gelu(xe @ w1[e]), bf16 out -------------------------
// Block tile 128(M) x 64(N), 8 waves, 32x32 per wave -> 4 WMMA / wave / K-step.
__global__ __launch_bounds__(256)
void moe_ffn1(const unsigned short* __restrict__ xe, const unsigned short* __restrict__ w1t,
              unsigned short* __restrict__ h) {
    __shared__ __align__(16) unsigned short lA[2][128 * LDA];
    __shared__ __align__(16) unsigned short lB[2][64 * LDA];
    int bin = blockIdx.z, e = bin & 7;
    int m0 = blockIdx.y * 128, n0 = blockIdx.x * 64;
    const unsigned short* A = xe + (size_t)bin * CAP * HS;  // [512][1024] bf16
    const unsigned short* B = w1t + (size_t)e * FFN * HS;   // [4096][1024] bf16 (N-major)
    int tid = threadIdx.x, lane = tid & 31, wid = tid >> 5;
    int mw = wid & 3, nw = wid >> 2;
    int l16 = lane & 15, hi = lane >> 4;
    int k0 = hi * 8, k1 = 16 + hi * 8;
    int srow = tid >> 2, skc = (tid & 3) * 8; // staging: 16B chunk per thread
    v8f acc00 = {}, acc01 = {}, acc10 = {}, acc11 = {};

    cp16_g2l(&lA[0][srow * LDA + skc],        A + (size_t)(m0 + srow) * HS + skc);
    cp16_g2l(&lA[0][(srow + 64) * LDA + skc], A + (size_t)(m0 + srow + 64) * HS + skc);
    cp16_g2l(&lB[0][srow * LDA + skc],        B + (size_t)(n0 + srow) * HS + skc);

    int buf = 0;
    for (int kk = 0; kk < HS; kk += 32) {
        if (kk + 32 < HS) {
            int nb = buf ^ 1, kn = kk + 32;
            cp16_g2l(&lA[nb][srow * LDA + skc],        A + (size_t)(m0 + srow) * HS + kn + skc);
            cp16_g2l(&lA[nb][(srow + 64) * LDA + skc], A + (size_t)(m0 + srow + 64) * HS + kn + skc);
            cp16_g2l(&lB[nb][srow * LDA + skc],        B + (size_t)(n0 + srow) * HS + kn + skc);
            wait_async<3>();
        } else {
            wait_async<0>();
        }
        __syncthreads();
        const unsigned short* cA = lA[buf];
        const unsigned short* cB = lB[buf];
        v16bf a0 = make_frag(cA, mw * 32 + l16,      k0, k1);
        v16bf a1 = make_frag(cA, mw * 32 + 16 + l16, k0, k1);
        v16bf b0 = make_frag(cB, nw * 32 + l16,      k0, k1);
        v16bf b1 = make_frag(cB, nw * 32 + 16 + l16, k0, k1);
        acc00 = wmma_bf16(a0, b0, acc00);
        acc01 = wmma_bf16(a0, b1, acc01);
        acc10 = wmma_bf16(a1, b0, acc10);
        acc11 = wmma_bf16(a1, b1, acc11);
        __syncthreads();
        buf ^= 1;
    }
    int colA = n0 + nw * 32 + l16, colB = colA + 16;
#pragma unroll
    for (int r = 0; r < 8; ++r) {
        int r0 = mw * 32 + r + 8 * hi, r1 = r0 + 16;
        size_t o0 = ((size_t)bin * CAP + m0 + r0) * FFN;
        size_t o1 = ((size_t)bin * CAP + m0 + r1) * FFN;
        h[o0 + colA] = f2bfu(gelu_tanh(acc00[r]));
        h[o0 + colB] = f2bfu(gelu_tanh(acc01[r]));
        h[o1 + colA] = f2bfu(gelu_tanh(acc10[r]));
        h[o1 + colB] = f2bfu(gelu_tanh(acc11[r]));
    }
}

// ---------------- 6) y = h @ w2[e]; gate-weighted scatter-add ---------------
__global__ __launch_bounds__(256)
void moe_ffn2(const unsigned short* __restrict__ hbuf, const unsigned short* __restrict__ w2t,
              const int* __restrict__ bins, const int* __restrict__ counts,
              const float* __restrict__ topw, float* __restrict__ out) {
    __shared__ __align__(16) unsigned short lA[2][128 * LDA];
    __shared__ __align__(16) unsigned short lB[2][64 * LDA];
    int bin = blockIdx.z, e = bin & 7, kslot = bin >> 3;
    int m0 = blockIdx.y * 128, n0 = blockIdx.x * 64;
    const unsigned short* A = hbuf + (size_t)bin * CAP * FFN; // [512][4096] bf16
    const unsigned short* B = w2t + (size_t)e * HS * FFN;     // [1024][4096] bf16 (N-major)
    int tid = threadIdx.x, lane = tid & 31, wid = tid >> 5;
    int mw = wid & 3, nw = wid >> 2;
    int l16 = lane & 15, hi = lane >> 4;
    int k0 = hi * 8, k1 = 16 + hi * 8;
    int srow = tid >> 2, skc = (tid & 3) * 8;
    v8f acc00 = {}, acc01 = {}, acc10 = {}, acc11 = {};

    cp16_g2l(&lA[0][srow * LDA + skc],        A + (size_t)(m0 + srow) * FFN + skc);
    cp16_g2l(&lA[0][(srow + 64) * LDA + skc], A + (size_t)(m0 + srow + 64) * FFN + skc);
    cp16_g2l(&lB[0][srow * LDA + skc],        B + (size_t)(n0 + srow) * FFN + skc);

    int buf = 0;
    for (int kk = 0; kk < FFN; kk += 32) {
        if (kk + 32 < FFN) {
            int nb = buf ^ 1, kn = kk + 32;
            cp16_g2l(&lA[nb][srow * LDA + skc],        A + (size_t)(m0 + srow) * FFN + kn + skc);
            cp16_g2l(&lA[nb][(srow + 64) * LDA + skc], A + (size_t)(m0 + srow + 64) * FFN + kn + skc);
            cp16_g2l(&lB[nb][srow * LDA + skc],        B + (size_t)(n0 + srow) * FFN + kn + skc);
            wait_async<3>();
        } else {
            wait_async<0>();
        }
        __syncthreads();
        const unsigned short* cA = lA[buf];
        const unsigned short* cB = lB[buf];
        v16bf a0 = make_frag(cA, mw * 32 + l16,      k0, k1);
        v16bf a1 = make_frag(cA, mw * 32 + 16 + l16, k0, k1);
        v16bf b0 = make_frag(cB, nw * 32 + l16,      k0, k1);
        v16bf b1 = make_frag(cB, nw * 32 + 16 + l16, k0, k1);
        acc00 = wmma_bf16(a0, b0, acc00);
        acc01 = wmma_bf16(a0, b1, acc01);
        acc10 = wmma_bf16(a1, b0, acc10);
        acc11 = wmma_bf16(a1, b1, acc11);
        __syncthreads();
        buf ^= 1;
    }
    int cnt = counts[bin];
    int colA = n0 + nw * 32 + l16, colB = colA + 16;
#pragma unroll
    for (int r = 0; r < 8; ++r) {
        int s0 = m0 + mw * 32 + r + 8 * hi, s1 = s0 + 16;
        if (s0 < cnt) {
            int tok = bins[bin * CAP + s0];
            float wgt = topw[tok * TOPK + kslot];
            atomicAdd(&out[(size_t)tok * HS + colA], wgt * acc00[r]);
            atomicAdd(&out[(size_t)tok * HS + colB], wgt * acc01[r]);
        }
        if (s1 < cnt) {
            int tok = bins[bin * CAP + s1];
            float wgt = topw[tok * TOPK + kslot];
            atomicAdd(&out[(size_t)tok * HS + colA], wgt * acc10[r]);
            atomicAdd(&out[(size_t)tok * HS + colB], wgt * acc11[r]);
        }
    }
}

extern "C" void kernel_launch(void* const* d_in, const int* in_sizes, int n_in,
                              void* d_out, int out_size, void* d_ws, size_t ws_size,
                              hipStream_t stream) {
    (void)in_sizes; (void)n_in; (void)out_size; (void)ws_size;
    const float* x        = (const float*)d_in[0]; // [4096][1024]
    const float* router_w = (const float*)d_in[1]; // [1024][8]
    const float* w1       = (const float*)d_in[2]; // [8][1024][4096]
    const float* w2       = (const float*)d_in[3]; // [8][4096][1024]
    const float* bias     = (const float*)d_in[4]; // [1024]
    float* out = (float*)d_out;

    char* ws = (char*)d_ws;
    size_t off = 0;
    float* topw = (float*)(ws + off); off += (size_t)TOKENS * TOPK * 4;
    int*   tope = (int*)(ws + off);   off += (size_t)TOKENS * TOPK * 4;
    int* counts = (int*)(ws + off);   off += 256;
    int*   bins = (int*)(ws + off);   off += (size_t)NBINS * CAP * 4;
    off = (off + 255) & ~(size_t)255;
    unsigned short* xe   = (unsigned short*)(ws + off); off += (size_t)NBINS * CAP * HS * 2;
    off = (off + 255) & ~(size_t)255;
    unsigned short* hbuf = (unsigned short*)(ws + off); off += (size_t)NBINS * CAP * FFN * 2;
    off = (off + 255) & ~(size_t)255;
    unsigned short* w1t  = (unsigned short*)(ws + off); off += (size_t)NUM_EXPERTS * HS * FFN * 2;
    off = (off + 255) & ~(size_t)255;
    unsigned short* w2t  = (unsigned short*)(ws + off); off += (size_t)NUM_EXPERTS * FFN * HS * 2;

    // Weight transpose+convert: w1 [K=HS][N=FFN] -> w1t [FFN][HS] bf16, etc.
    transpose_cvt<<<dim3(FFN / 64, HS / 64, NUM_EXPERTS), 256, 0, stream>>>(w1, w1t, HS, FFN);
    transpose_cvt<<<dim3(HS / 64, FFN / 64, NUM_EXPERTS), 256, 0, stream>>>(w2, w2t, FFN, HS);

    moe_router   <<<TOKENS / 8, 256, 0, stream>>>(x, router_w, topw, tope);
    moe_bin      <<<NBINS, 32, 0, stream>>>(tope, bins, counts);
    moe_gather   <<<NBINS * CAP, 256, 0, stream>>>(x, bins, counts, xe);
    moe_bias_init<<<(TOKENS * HS) / 256, 256, 0, stream>>>(bias, out);
    moe_ffn1     <<<dim3(FFN / 64, CAP / 128, NBINS), 256, 0, stream>>>(xe, w1t, hbuf);
    moe_ffn2     <<<dim3(HS / 64, CAP / 128, NBINS), 256, 0, stream>>>(hbuf, w2t, bins, counts, topw, out);
}
